// QCL_classifier_79826262163577
// MI455X (gfx1250) — matrix operand order repaired and checked by reading, not compiled, not run on hardware
//
#include <hip/hip_runtime.h>
#include <hip/hip_bf16.h>
#include <math.h>
#include <stdint.h>

#define N_QUBITS 10
#define DIM      1024
#define DEPTH    5
#define BATCH    16384
#define IN_DIM   784
#define KPAD     800      // 25 * 32 (K padded for uniform WMMA K-loop)
#define NDIM     2048     // re/im interleaved: n = 2*a + comp
#define TBM      128
#define TBN      128
#define TBK      32
#define KSTEPS   (KPAD / TBK)

typedef __attribute__((ext_vector_type(16))) __bf16 v16bf;
typedef __attribute__((ext_vector_type(8)))  float  v8f;

// float -> bf16 round-to-nearest-even
__device__ __forceinline__ unsigned short f2bf(float f) {
  unsigned u = __float_as_uint(f);
  unsigned r = u + 0x7FFFu + ((u >> 16) & 1u);
  return (unsigned short)(r >> 16);
}

// ---------------------------------------------------------------------------
// Kernel 0: zero the group-sum accumulator and the output scalar
// ---------------------------------------------------------------------------
__global__ void k_init(float* __restrict__ S, float* __restrict__ out) {
  int i = blockIdx.x * blockDim.x + threadIdx.x;
  if (i < BATCH * 4) S[i] = 0.f;
  if (i == 0) out[0] = 0.f;
}

// ---------------------------------------------------------------------------
// Kernel 1: build bf16 B-matrix Bm[n][k], n = 2*a + (0:re,1:im), k = input col.
// Column k of the circuit unitary = circuit applied to basis state e_k.
// Columns 784..799 are zero padding. One block per column; state in LDS.
// ---------------------------------------------------------------------------
__global__ __launch_bounds__(256) void k_build(const float* __restrict__ w,
                                               unsigned short* __restrict__ Bm) {
  __shared__ float sre[DIM];
  __shared__ float sim_[DIM];
  const int c = blockIdx.x;   // 0..799
  const int t = threadIdx.x;  // 0..255

  if (c < IN_DIM) {
    for (int i = t; i < DIM; i += 256) { sre[i] = (i == c) ? 1.f : 0.f; sim_[i] = 0.f; }
    __syncthreads();

    for (int d = 0; d < DEPTH; ++d) {
      // CNOT ring: new[m] = old[T0(T1(...T9(m)))]; qubit q <-> bit (9-q)
      float rr[4], ii[4];
      #pragma unroll
      for (int e = 0; e < 4; ++e) {
        int m = t + 256 * e;
        int idx = m;
        #pragma unroll
        for (int i = N_QUBITS - 1; i >= 0; --i) {
          int cb = (N_QUBITS - 1) - i;
          int tb = (N_QUBITS - 1) - ((i + 1) % N_QUBITS);
          if ((idx >> cb) & 1) idx ^= (1 << tb);
        }
        rr[e] = sre[idx]; ii[e] = sim_[idx];
      }
      __syncthreads();
      #pragma unroll
      for (int e = 0; e < 4; ++e) { int m = t + 256 * e; sre[m] = rr[e]; sim_[m] = ii[e]; }
      __syncthreads();

      // per-qubit Rot: RZ(omega) RY(theta) RZ(phi)
      for (int q = 0; q < N_QUBITS; ++q) {
        const float* wg = w + ((d * N_QUBITS) + q) * 3;
        float phi = wg[0], th = wg[1], om = wg[2];
        float ca = cosf(0.5f * th), sa = sinf(0.5f * th);
        float A = 0.5f * (phi + om), B = 0.5f * (phi - om);
        float sA, cA, sB, cB;
        sincosf(A, &sA, &cA);
        sincosf(B, &sB, &cB);
        float m00r =  cA * ca, m00i = -sA * ca;
        float m11r =  cA * ca, m11i =  sA * ca;
        float m01r = -cB * sa, m01i = -sB * sa;
        float m10r =  cB * sa, m10i = -sB * sa;
        int b = (N_QUBITS - 1) - q;
        int low = (1 << b) - 1;
        #pragma unroll
        for (int e = 0; e < 2; ++e) {
          int p = t + 256 * e;  // pair index 0..511
          int j0 = ((p & ~low) << 1) | (p & low);
          int j1 = j0 | (1 << b);
          float s0r = sre[j0], s0i = sim_[j0];
          float s1r = sre[j1], s1i = sim_[j1];
          sre[j0]  = m00r * s0r - m00i * s0i + m01r * s1r - m01i * s1i;
          sim_[j0] = m00r * s0i + m00i * s0r + m01r * s1i + m01i * s1r;
          sre[j1]  = m10r * s0r - m10i * s0i + m11r * s1r - m11i * s1i;
          sim_[j1] = m10r * s0i + m10i * s0r + m11r * s1i + m11i * s1r;
        }
        __syncthreads();
      }
    }
  } else {
    for (int i = t; i < DIM; i += 256) { sre[i] = 0.f; sim_[i] = 0.f; }
    __syncthreads();
  }

  for (int a = t; a < DIM; a += 256) {
    Bm[(size_t)(2 * a)     * KPAD + c] = f2bf(sre[a]);
    Bm[(size_t)(2 * a + 1) * KPAD + c] = f2bf(sim_[a]);
  }
}

// ---------------------------------------------------------------------------
// Kernel 2: bf16 WMMA GEMM [16384 x 800] x [800 x 2048] with squared-group-sum
// epilogue. Block tile 128(batch) x 128(n); 8 waves in 4x2; wave tile 32x64.
// All n in one block belong to a single qubit-(0,1) group g = blockIdx.x>>2.
// B tile is staged with GLOBAL_LOAD_ASYNC_TO_LDS_B128 (ASYNCcnt path).
// ---------------------------------------------------------------------------
__global__ __launch_bounds__(256) void k_gemm(const float* __restrict__ X,
                                              const unsigned short* __restrict__ Bm,
                                              float* __restrict__ S) {
  // As: [row][k] with K in natural order (half hf holds k = 16*hf + s), so each
  // lane's 16 A-fragment values are one contiguous 32B chunk.
  __shared__ alignas(64) unsigned short As[TBM * TBK];
  __shared__ alignas(64) unsigned short Bs[TBN * TBK];  // [n][k 0..31]

  const int t    = threadIdx.x;
  const int lane = t & 31;
  const int wv   = t >> 5;
  const int wb   = wv >> 1;   // batch sub 0..3
  const int wn   = wv & 1;    // n sub 0..1
  const int b0   = blockIdx.y * TBM;
  const int n0   = blockIdx.x * TBN;
  const int g    = blockIdx.x >> 2;   // 64 amplitudes/block, 256/group

  v8f acc[2][4];
  #pragma unroll
  for (int i = 0; i < 2; ++i)
    #pragma unroll
    for (int j = 0; j < 4; ++j)
      #pragma unroll
      for (int e = 0; e < 8; ++e) acc[i][j][e] = 0.f;

  const int hf = lane >> 4;   // half-wave
  const int lr = lane & 15;

  // per-thread B-tile assignment (row t>>1, 32B chunk (t&1)*16 elements)
  const int brow = t >> 1;
  const int bco  = (t & 1) * 16;
  // generic->LDS: low 32 bits of a flat pointer into LDS are the LDS address
  const unsigned bs_lds = (unsigned)(uintptr_t)(&Bs[brow * TBK + bco]);

  for (int kt = 0; kt < KSTEPS; ++kt) {
    // ---- stage B tile via async DMA to LDS (no VGPR data movement) ----
    // INST_OFFSET is added to both LDS and global addresses, so one address
    // pair covers both 16B chunks of this thread's 32B row segment.
    {
      const unsigned short* gp = Bm + (size_t)(brow + n0) * KPAD + kt * TBK + bco;
      asm volatile(
          "global_load_async_to_lds_b128 %0, %1, off\n\t"
          "global_load_async_to_lds_b128 %0, %1, off offset:16"
          :: "v"(bs_lds), "v"(gp) : "memory");
    }
    // ---- stage A (f32 -> bf16, K in natural order, as in Bs) ----
    {
      int r   = t >> 3;         // 0..31
      int c0  = (t & 7) * 4;    // 0..28
      int gg  = c0 >> 3;
      int g2  = gg;
      int pos = (c0 & 7) + (g2 << 3);
      #pragma unroll
      for (int pass = 0; pass < 4; ++pass) {
        int row = r + pass * 32;
        unsigned short o0, o1, o2, o3;
        if (kt == KSTEPS - 1 && c0 >= 16) { o0 = o1 = o2 = o3 = 0; }
        else {
          const float* gp = X + (size_t)(b0 + row) * IN_DIM + kt * TBK + c0;
          const float4 v = *(const float4*)gp;
          if (kt + 1 < KSTEPS) __builtin_prefetch(gp + TBK, 0, 1);
          o0 = f2bf(v.x); o1 = f2bf(v.y); o2 = f2bf(v.z); o3 = f2bf(v.w);
        }
        unsigned short* d = &As[row * TBK + pos];
        d[0] = o0; d[1] = o1; d[2] = o2; d[3] = o3;
      }
    }
    // all of this wave's async B transfers must land before the barrier
    asm volatile("s_wait_asynccnt 0x0" ::: "memory");
    __syncthreads();

    v16bf a[2], bf[4];
    #pragma unroll
    for (int rb = 0; rb < 2; ++rb)
      a[rb] = *(const v16bf*)&As[(wb * 32 + rb * 16 + lr) * TBK + hf * 16];
    #pragma unroll
    for (int j = 0; j < 4; ++j)
      bf[j] = *(const v16bf*)&Bs[(wn * 64 + j * 16 + lr) * TBK + hf * 16];

    #pragma unroll
    for (int rb = 0; rb < 2; ++rb)
      #pragma unroll
      for (int j = 0; j < 4; ++j)
        acc[rb][j] = __builtin_amdgcn_wmma_f32_16x16x32_bf16(
            false, a[rb], false, bf[j], (short)0, acc[rb][j], false, false);

    __syncthreads();
  }

  // Epilogue: sum of squares over this wave's 64 n-columns per batch row.
  // C layout: VGPR e, lane l -> M = e + 8*(l>>4), N = l&15.
  #pragma unroll
  for (int rb = 0; rb < 2; ++rb) {
    float sq[8];
    #pragma unroll
    for (int e = 0; e < 8; ++e) {
      float s = 0.f;
      #pragma unroll
      for (int j = 0; j < 4; ++j) { float v = acc[rb][j][e]; s += v * v; }
      sq[e] = s;
    }
    #pragma unroll
    for (int e = 0; e < 8; ++e) {
      float v = sq[e];
      v += __shfl_xor(v, 1, 32);
      v += __shfl_xor(v, 2, 32);
      v += __shfl_xor(v, 4, 32);
      v += __shfl_xor(v, 8, 32);
      if ((lane & 15) == e) {
        int row = b0 + wb * 32 + rb * 16 + e + 8 * hf;
        atomicAdd(&S[row * 4 + g], v);
      }
    }
  }
}

// ---------------------------------------------------------------------------
// Kernel 3: z0/z1 from group sums (normalization folded in), NLL loss, mean.
// ---------------------------------------------------------------------------
__global__ __launch_bounds__(256) void k_loss(const float* __restrict__ S,
                                              const int* __restrict__ y,
                                              float* __restrict__ out) {
  __shared__ float red[256];
  int b = blockIdx.x * 256 + threadIdx.x;
  float s0 = S[b * 4 + 0], s1 = S[b * 4 + 1], s2 = S[b * 4 + 2], s3 = S[b * 4 + 3];
  float inv = 1.f / (s0 + s1 + s2 + s3);
  float z0 = (s0 + s1 - s2 - s3) * inv;   // qubit 0 (MSB group bit)
  float z1 = (s0 - s1 + s2 - s3) * inv;   // qubit 1
  float m = fmaxf(z0, z1);
  float lse = m + logf(expf(z0 - m) + expf(z1 - m));
  float zy = (y[b] == 0) ? z0 : z1;
  red[threadIdx.x] = (lse - zy) * (1.0f / BATCH);
  __syncthreads();
  for (int st = 128; st > 0; st >>= 1) {
    if (threadIdx.x < st) red[threadIdx.x] += red[threadIdx.x + st];
    __syncthreads();
  }
  if (threadIdx.x == 0) atomicAdd(out, red[0]);
}

extern "C" void kernel_launch(void* const* d_in, const int* in_sizes, int n_in,
                              void* d_out, int out_size, void* d_ws, size_t ws_size,
                              hipStream_t stream) {
  const float* x = (const float*)d_in[0];
  const float* w = (const float*)d_in[1];
  const int*   y = (const int*)d_in[2];
  float* out = (float*)d_out;

  unsigned short* Bm = (unsigned short*)d_ws;                       // 2048*800 bf16 = 3.125 MB
  float* S = (float*)((char*)d_ws + (size_t)NDIM * KPAD * 2);       // 16384*4 f32 = 256 KB

  k_init<<<(BATCH * 4 + 255) / 256, 256, 0, stream>>>(S, out);
  k_build<<<KPAD, 256, 0, stream>>>(w, Bm);
  dim3 grid(NDIM / TBN, BATCH / TBM);
  k_gemm<<<grid, 256, 0, stream>>>(x, Bm, S);
  k_loss<<<BATCH / 256, 256, 0, stream>>>(S, y, out);
}
